// QLoRABigNet_74165495267603
// MI455X (gfx1250) — matrix operand order repaired and checked by reading, not compiled, not run on hardware
//
#include <hip/hip_runtime.h>

#define DIM   1024
#define NROWS 16384
#define RANK  32
#define BM    64
#define BN    128
#define BK    32
#define LW    40   // LDS row stride in ushorts: 80B = 20 banks -> conflict-free, 16B aligned rows

#if __has_builtin(__builtin_amdgcn_global_load_async_to_lds_b128) && \
    __has_builtin(__builtin_amdgcn_s_wait_asynccnt)
#define HAVE_ASYNC_LDS 1
#else
#define HAVE_ASYNC_LDS 0
#endif

typedef __attribute__((ext_vector_type(16))) __bf16 v16bf;
typedef __attribute__((ext_vector_type(8)))  float  v8f;
typedef __attribute__((ext_vector_type(4)))  int    v4i;

union Frag { v16bf v; unsigned u[8]; };

__device__ __forceinline__ unsigned short f2bf(float x) {
  unsigned u = __float_as_uint(x);
  u = u + 0x7fffu + ((u >> 16) & 1u);   // round-to-nearest-even
  return (unsigned short)(u >> 16);
}
__device__ __forceinline__ float bf2f(unsigned short h) {
  return __uint_as_float(((unsigned)h) << 16);
}
__device__ __forceinline__ void split2(float a, float b, unsigned &uh, unsigned &ul) {
  unsigned short ah = f2bf(a), bh = f2bf(b);
  unsigned short al = f2bf(a - bf2f(ah)), bl = f2bf(b - bf2f(bh));
  uh = (unsigned)ah | ((unsigned)bh << 16);
  ul = (unsigned)al | ((unsigned)bl << 16);
}
__device__ __forceinline__ v8f wmma_bf16(const Frag &a, const Frag &b, v8f c) {
  return __builtin_amdgcn_wmma_f32_16x16x32_bf16(false, a.v, false, b.v,
                                                 (short)0, c, false, false);
}

#if HAVE_ASYNC_LDS
__device__ __forceinline__ void async_copy16(const void *g, void *l) {
  __builtin_amdgcn_global_load_async_to_lds_b128((v4i *)g, (v4i *)l, 0, 0);
}
#endif

// ------- streaming fp32 -> packed bf16 hi/lo split (8 elems / thread) ---------
__global__ __launch_bounds__(256)
void k_split(const float *__restrict__ in, uint4 *__restrict__ OH,
             uint4 *__restrict__ OL) {
  size_t gid = (size_t)blockIdx.x * 256 + threadIdx.x;
  const float4 *s = (const float4 *)in;
  float4 a = s[2 * gid], b = s[2 * gid + 1];
  uint4 uh, ul;
  split2(a.x, a.y, uh.x, ul.x);
  split2(a.z, a.w, uh.y, ul.y);
  split2(b.x, b.y, uh.z, ul.z);
  split2(b.z, b.w, uh.w, ul.w);
  OH[gid] = uh;
  OL[gid] = ul;
}

// ---------------- dequant: q4 -> W hi/lo bf16, row-major [out][in] ------------
__global__ __launch_bounds__(256)
void k_dequant(const int *__restrict__ q4, const float *__restrict__ norm,
               unsigned *__restrict__ WH, unsigned *__restrict__ WL) {
  int g = blockIdx.x * 256 + threadIdx.x;           // one 16-elem group / thread
  const int *q = q4 + (size_t)g * 8;
  float nr = norm[g];
  #pragma unroll
  for (int j = 0; j < 8; ++j) {
    int qq = q[j];
    float w0 = ((float)(qq & 15)        * (2.0f / 15.0f) - 1.0f) * nr;
    float w1 = ((float)((qq >> 4) & 15) * (2.0f / 15.0f) - 1.0f) * nr;
    unsigned uh, ul;
    split2(w0, w1, uh, ul);
    WH[(size_t)g * 8 + j] = uh;
    WL[(size_t)g * 8 + j] = ul;
  }
}

// ---------------- T = x @ la^T  (M=16384, N=32, K=1024), pre-split inputs -----
__global__ __launch_bounds__(256)
void k_lora_t(const unsigned *__restrict__ XH, const unsigned *__restrict__ XL,
              const unsigned *__restrict__ LAH, const unsigned *__restrict__ LAL,
              float *__restrict__ T) {
  const int t = threadIdx.x, lane = t & 31, w = t >> 5;
  const int half = lane >> 4, l15 = lane & 15;
  const int m0 = blockIdx.x * 128 + w * 16;
  const int m = m0 + l15;
  const unsigned *xh = XH + (size_t)m * (DIM / 2);
  const unsigned *xl = XL + (size_t)m * (DIM / 2);
  v8f acc0 = {}, acc1 = {};
  for (int k0 = 0; k0 < DIM; k0 += 32) {
    int kd = k0 >> 1;
    Frag aH, aL;
    #pragma unroll
    for (int r = 0; r < 8; ++r) {
      int off = (r < 4) ? (4 * half + r) : (8 + 4 * half + (r - 4));
      aH.u[r] = xh[kd + off];
      aL.u[r] = xl[kd + off];
    }
    #pragma unroll
    for (int nt = 0; nt < 2; ++nt) {
      int n = nt * 16 + l15;
      const unsigned *ah = LAH + (size_t)n * (DIM / 2);
      const unsigned *al = LAL + (size_t)n * (DIM / 2);
      Frag bH, bL;
      #pragma unroll
      for (int r = 0; r < 8; ++r) {
        bH.u[r] = ah[kd + ((r < 4) ? (4 * half + r) : (8 + 4 * half + (r - 4)))];
        bL.u[r] = al[kd + ((r < 4) ? (4 * half + r) : (8 + 4 * half + (r - 4)))];
      }
      v8f &acc = nt ? acc1 : acc0;
      acc = wmma_bf16(aH, bH, acc);
      acc = wmma_bf16(aH, bL, acc);
      acc = wmma_bf16(aL, bH, acc);
    }
  }
  #pragma unroll
  for (int v = 0; v < 8; ++v) {
    int mr = m0 + v + 8 * half;
    T[(size_t)mr * RANK + l15]      = acc0[v];
    T[(size_t)mr * RANK + 16 + l15] = acc1[v];
  }
}

// -- out = maybe_relu(X @ W^T + bias + T @ lb^T); X/W pre-split bf16 hi/lo -----
__global__ __launch_bounds__(256)
void k_gemm(const unsigned short *__restrict__ XH,
            const unsigned short *__restrict__ XL,
            const unsigned short *__restrict__ WH,
            const unsigned short *__restrict__ WL,
            const float *__restrict__ T,
            const unsigned *__restrict__ LBH, const unsigned *__restrict__ LBL,
            const float *__restrict__ bias, float *__restrict__ out, int relu) {
  __shared__ unsigned short AsH[BM][LW], AsL[BM][LW];
  __shared__ unsigned short BsH[BN][LW], BsL[BN][LW];
  const int t = threadIdx.x, lane = t & 31, w = t >> 5;
  const int half = lane >> 4, l15 = lane & 15;
  const int wm = (w >> 2) * 32, wn = (w & 3) * 32;
  const int n0 = blockIdx.x * BN, m0 = blockIdx.y * BM;

  v8f zero = {};
  v8f acc[2][2];
  acc[0][0] = zero; acc[0][1] = zero; acc[1][0] = zero; acc[1][1] = zero;

  const int arow = t >> 2, acol = (t & 3) * 8;   // A: 64 rows x 32 cols, 8 elems/thread
  const int brow = t >> 1, bcol = (t & 1) * 16;  // B: 128 rows x 32 cols, 16 elems/thread

  for (int k0 = 0; k0 < DIM; k0 += BK) {
    size_t aoff = (size_t)(m0 + arow) * DIM + k0 + acol;
    size_t boff = (size_t)(n0 + brow) * DIM + k0 + bcol;
#if HAVE_ASYNC_LDS
    async_copy16(XH + aoff, &AsH[arow][acol]);
    async_copy16(XL + aoff, &AsL[arow][acol]);
    async_copy16(WH + boff,     &BsH[brow][bcol]);
    async_copy16(WH + boff + 8, &BsH[brow][bcol + 8]);
    async_copy16(WL + boff,     &BsL[brow][bcol]);
    async_copy16(WL + boff + 8, &BsL[brow][bcol + 8]);
    __builtin_amdgcn_s_wait_asynccnt(0);
#else
    *(uint4 *)&AsH[arow][acol] = *(const uint4 *)(XH + aoff);
    *(uint4 *)&AsL[arow][acol] = *(const uint4 *)(XL + aoff);
    *(uint4 *)&BsH[brow][bcol]     = *(const uint4 *)(WH + boff);
    *(uint4 *)&BsH[brow][bcol + 8] = *(const uint4 *)(WH + boff + 8);
    *(uint4 *)&BsL[brow][bcol]     = *(const uint4 *)(WL + boff);
    *(uint4 *)&BsL[brow][bcol + 8] = *(const uint4 *)(WL + boff + 8);
#endif
    __syncthreads();

    Frag aH[2], aL[2], bH[2], bL[2];
    #pragma unroll
    for (int mt = 0; mt < 2; ++mt) {
      int m = wm + mt * 16 + l15;
      const unsigned *ph = (const unsigned *)&AsH[m][0];
      const unsigned *pl = (const unsigned *)&AsL[m][0];
      int hb = 4 * half;
      #pragma unroll
      for (int r = 0; r < 4; ++r) {
        aH[mt].u[r] = ph[hb + r]; aH[mt].u[4 + r] = ph[8 + hb + r];
        aL[mt].u[r] = pl[hb + r]; aL[mt].u[4 + r] = pl[8 + hb + r];
      }
    }
    #pragma unroll
    for (int nt = 0; nt < 2; ++nt) {
      int n = wn + nt * 16 + l15;
      const unsigned *ph = (const unsigned *)&BsH[n][0];
      const unsigned *pl = (const unsigned *)&BsL[n][0];
      #pragma unroll
      for (int r = 0; r < 8; ++r) {
        bH[nt].u[r] = ph[(r < 4) ? (4 * half + r) : (8 + 4 * half + (r - 4))];
        bL[nt].u[r] = pl[(r < 4) ? (4 * half + r) : (8 + 4 * half + (r - 4))];
      }
    }
    #pragma unroll
    for (int mt = 0; mt < 2; ++mt)
      #pragma unroll
      for (int nt = 0; nt < 2; ++nt) {
        acc[mt][nt] = wmma_bf16(aH[mt], bH[nt], acc[mt][nt]);
        acc[mt][nt] = wmma_bf16(aH[mt], bL[nt], acc[mt][nt]);
        acc[mt][nt] = wmma_bf16(aL[mt], bH[nt], acc[mt][nt]);
      }
    __syncthreads();
  }

  // epilogue: + T @ lb^T (rank 32 == one K=32 WMMA triple), + bias, relu, store
  #pragma unroll
  for (int mt = 0; mt < 2; ++mt) {
    int m = m0 + wm + mt * 16 + l15;
    Frag tH, tL;
    #pragma unroll
    for (int r = 0; r < 8; ++r) {
      int k = (r < 4) ? (8 * half + 2 * r) : (16 + 8 * half + 2 * (r - 4));
      split2(T[(size_t)m * RANK + k], T[(size_t)m * RANK + k + 1], tH.u[r], tL.u[r]);
    }
    #pragma unroll
    for (int nt = 0; nt < 2; ++nt) {
      int n = n0 + wn + nt * 16 + l15;
      const unsigned *lbh = LBH + (size_t)n * (RANK / 2);
      const unsigned *lbl = LBL + (size_t)n * (RANK / 2);
      Frag lH, lL;
      #pragma unroll
      for (int r = 0; r < 8; ++r) {
        lH.u[r] = lbh[(r < 4) ? (4 * half + r) : (8 + 4 * half + (r - 4))];
        lL.u[r] = lbl[(r < 4) ? (4 * half + r) : (8 + 4 * half + (r - 4))];
      }
      v8f c = acc[mt][nt];
      c = wmma_bf16(tH, lH, c);
      c = wmma_bf16(tH, lL, c);
      c = wmma_bf16(tL, lH, c);
      float bn = bias[n];
      #pragma unroll
      for (int vv = 0; vv < 8; ++vv) {
        float y = c[vv] + bn;
        if (relu) y = fmaxf(y, 0.0f);
        out[(size_t)(m0 + wm + mt * 16 + vv + 8 * half) * DIM + n] = y;
      }
    }
  }
}

// ---------------- residual add + (optional) LayerNorm, one row per block ------
__global__ __launch_bounds__(256)
void k_resln(const float *__restrict__ r, const float *__restrict__ hin,
             float *__restrict__ hout, const float *__restrict__ gamma,
             const float *__restrict__ beta, int doln) {
  __shared__ float sbuf[256];
  const int row = blockIdx.x, t = threadIdx.x;
  float v[4];
  float s = 0.f, sq = 0.f;
  #pragma unroll
  for (int j = 0; j < 4; ++j) {
    int idx = t + j * 256;
    float x = r[(size_t)row * DIM + idx] + hin[(size_t)row * DIM + idx];
    v[j] = x; s += x; sq += x * x;
  }
  if (!doln) {
    #pragma unroll
    for (int j = 0; j < 4; ++j) hout[(size_t)row * DIM + t + j * 256] = v[j];
    return;
  }
  sbuf[t] = s; __syncthreads();
  for (int st = 128; st > 0; st >>= 1) { if (t < st) sbuf[t] += sbuf[t + st]; __syncthreads(); }
  float mu = sbuf[0] * (1.0f / DIM);
  __syncthreads();
  sbuf[t] = sq; __syncthreads();
  for (int st = 128; st > 0; st >>= 1) { if (t < st) sbuf[t] += sbuf[t + st]; __syncthreads(); }
  float var = sbuf[0] * (1.0f / DIM) - mu * mu;
  float inv = rsqrtf(var + 1e-5f);
  #pragma unroll
  for (int j = 0; j < 4; ++j) {
    int idx = t + j * 256;
    hout[(size_t)row * DIM + idx] = (v[j] - mu) * inv * gamma[idx] + beta[idx];
  }
}

extern "C" void kernel_launch(void *const *d_in, const int *in_sizes, int n_in,
                              void *d_out, int out_size, void *d_ws, size_t ws_size,
                              hipStream_t stream) {
  const float *x      = (const float *)d_in[0];
  const int   *w_q4   = (const int *)d_in[1];
  const float *w_norm = (const float *)d_in[2];
  const float *bias   = (const float *)d_in[3];
  const float *lora_a = (const float *)d_in[4];
  const float *lora_b = (const float *)d_in[5];
  const float *ln_g   = (const float *)d_in[6];
  const float *ln_b   = (const float *)d_in[7];
  float *out = (float *)d_out;

  const size_t NB = (size_t)NROWS * DIM * sizeof(float);   // 64 MB
  float *H = (float *)d_ws;
  float *R = (float *)((char *)d_ws + NB);
  unsigned short *XH = (unsigned short *)((char *)d_ws + 2 * NB);   // 32 MB
  unsigned short *XL = XH + (size_t)NROWS * DIM;                    // 32 MB
  float *T = (float *)((char *)d_ws + 3 * NB);                      // 2 MB
  unsigned *WHu  = (unsigned *)((char *)T + (size_t)NROWS * RANK * sizeof(float));
  unsigned *WLu  = WHu + (size_t)DIM * DIM / 2;
  unsigned *LAHu = WLu + (size_t)DIM * DIM / 2;
  unsigned *LALu = LAHu + (size_t)RANK * DIM / 2;
  unsigned *LBHu = LALu + (size_t)RANK * DIM / 2;
  unsigned *LBLu = LBHu + (size_t)DIM * RANK / 2;

  (void)hipMemcpyAsync(H, x, NB, hipMemcpyDeviceToDevice, stream);

  dim3 gGemm(DIM / BN, NROWS / BM);
  const int GROUPS  = DIM * DIM / 16;       // 65536 per layer
  const int GSPLITX = NROWS * DIM / 2048;   // 8192 blocks (8 elems/thread)
  const int GSPLITL = RANK * DIM / 2048;    // 16 blocks

  for (int blk = 0; blk < 6; ++blk) {
    const float *lin[3]  = {H, R, out};
    float       *lout[3] = {R, out, R};
    for (int j = 0; j < 3; ++j) {
      int i = blk * 3 + j;
      k_dequant<<<GROUPS / 256, 256, 0, stream>>>(
          w_q4 + (size_t)i * GROUPS * 8, w_norm + (size_t)i * GROUPS, WHu, WLu);
      k_split<<<GSPLITX, 256, 0, stream>>>(lin[j], (uint4 *)XH, (uint4 *)XL);
      k_split<<<GSPLITL, 256, 0, stream>>>(lora_a + (size_t)i * RANK * DIM,
                                           (uint4 *)LAHu, (uint4 *)LALu);
      k_split<<<GSPLITL, 256, 0, stream>>>(lora_b + (size_t)i * DIM * RANK,
                                           (uint4 *)LBHu, (uint4 *)LBLu);
      k_lora_t<<<NROWS / 128, 256, 0, stream>>>(
          (const unsigned *)XH, (const unsigned *)XL, LAHu, LALu, T);
      k_gemm<<<gGemm, 256, 0, stream>>>(
          XH, XL, (const unsigned short *)WHu, (const unsigned short *)WLu, T,
          LBHu, LBLu, bias + (size_t)i * DIM, lout[j], (j < 2) ? 1 : 0);
    }
    int last = (blk == 5);
    int lnidx = (blk < 5) ? blk : 4;
    k_resln<<<NROWS, 256, 0, stream>>>(R, H, last ? out : H,
        ln_g + (size_t)lnidx * DIM, ln_b + (size_t)lnidx * DIM, last ? 0 : 1);
  }
}
